// DeepIRT_29386166239698
// MI455X (gfx1250) — matrix operand order repaired and checked by reading, not compiled, not run on hardware
//
#include <hip/hip_runtime.h>
#include <hip/hip_bf16.h>
#include <math.h>

typedef __attribute__((ext_vector_type(16))) _Float16 v16h;
typedef __attribute__((ext_vector_type(8)))  _Float16 v8h;
typedef __attribute__((ext_vector_type(8)))  float    v8f;

#define B_   64
#define T_   500
#define BT_  (B_ * T_)
#define D_   64
#define S_   50
#define NC_  1024

// half-element counts in the f16 workspace region
#define EKH_SZ   (NC_ * D_)            // 65536
#define EVH_SZ   (2 * NC_ * D_)        // 131072
#define FRAG8_SZ (8 * 32 * 16)         // 4096 (4 ntiles x 2 kchunks)
#define FRAG16_SZ (16 * 32 * 16)       // 8192 (4 ntiles x 4 kchunks)

static __device__ __forceinline__ v8f wmma_f16(v16h a, v16h b, v8f c) {
    // D = A(16x32 f16) * B(32x16 f16) + C(16x16 f32)
    return __builtin_amdgcn_wmma_f32_16x16x32_f16(false, a, false, b, (short)0, c,
                                                  false, false);
}

static __device__ __forceinline__ v16h cat8(v8h lo, v8h hi) {
    return __builtin_shufflevector(lo, hi, 0, 1, 2, 3, 4, 5, 6, 7,
                                           8, 9, 10, 11, 12, 13, 14, 15);
}

// -------------------------------------------------------------------------
// Prep A: convert embedding tables to f16 once (1024+2048 rows, not 32000).
// -------------------------------------------------------------------------
__global__ __launch_bounds__(256)
void deepirt_cvt_tables(const float* __restrict__ ek, const float* __restrict__ ev,
                        _Float16* __restrict__ ek_h, _Float16* __restrict__ ev_h)
{
    const int idx = blockIdx.x * 256 + threadIdx.x;      // grid covers EKH+EVH
    if (idx < EKH_SZ) {
        ek_h[idx] = (_Float16)ek[idx];
    } else if (idx < EKH_SZ + EVH_SZ) {
        const int i2 = idx - EKH_SZ;
        ev_h[i2] = (_Float16)ev[i2];
    }
}

// -------------------------------------------------------------------------
// Prep B: pack Mk^T / We / Wa / Wf into WMMA B-fragment layout (f16):
//   dst[((nt*nchunks + c)*32 + lane)*16 + j] = B[K = c*32 + (j<8 ? 8*(lane>>4)+j : 16+8*(lane>>4)+j-8)]
//                                               [N = nt*16 + (lane&15)]
// Mk is stored (S x D) so B=Mk^T -> element = Mk[n][kk], zero-padded n>=50.
// -------------------------------------------------------------------------
__global__ __launch_bounds__(256)
void deepirt_pack_frags(const float* __restrict__ Mk, const float* __restrict__ We,
                        const float* __restrict__ Wa, const float* __restrict__ Wf,
                        _Float16* __restrict__ MkF, _Float16* __restrict__ WeF,
                        _Float16* __restrict__ WaF, _Float16* __restrict__ WfF)
{
    const int idx = blockIdx.x * 256 + threadIdx.x;      // 0 .. 20479
    _Float16* dst;
    const float* src;
    int ridx, nchunks, isMk = 0;
    if (idx < FRAG8_SZ)            { dst = MkF; src = Mk; ridx = idx;                nchunks = 2; isMk = 1; }
    else if (idx < 2 * FRAG8_SZ)   { dst = WeF; src = We; ridx = idx - FRAG8_SZ;     nchunks = 2; }
    else if (idx < 3 * FRAG8_SZ)   { dst = WaF; src = Wa; ridx = idx - 2 * FRAG8_SZ; nchunks = 2; }
    else if (idx < 3 * FRAG8_SZ + FRAG16_SZ)
                                   { dst = WfF; src = Wf; ridx = idx - 3 * FRAG8_SZ; nchunks = 4; }
    else return;

    const int frag = ridx >> 9;          // (lane,j) block of 512
    const int lane = (ridx >> 4) & 31;
    const int j    = ridx & 15;
    const int nt   = frag / nchunks;
    const int c    = frag % nchunks;
    const int m    = lane & 15;
    const int hi   = lane >> 4;
    const int n    = nt * 16 + m;
    const int kk   = c * 32 + (j < 8 ? hi * 8 + j : 16 + hi * 8 + (j - 8));

    float val;
    if (isMk) val = (n < S_) ? src[n * D_ + kk] : 0.f;   // B = Mk^T (with N pad)
    else      val = src[kk * D_ + n];
    dst[ridx] = (_Float16)val;
}

// -------------------------------------------------------------------------
// Phase 1: per 16-row tile (one wave32):
//   w = softmax(k @ Mk^T) ; e = sigmoid(v@We+be) ; a = tanh(v@Wa+ba)
// A fragments loaded directly from f16 tables; B fragments are one v16h load.
// -------------------------------------------------------------------------
__global__ __launch_bounds__(32)
void deepirt_phase1(const int*   __restrict__ concept_seq,
                    const int*   __restrict__ correct_seq,
                    const _Float16* __restrict__ ek_h,
                    const _Float16* __restrict__ ev_h,
                    const v16h* __restrict__ MkF,
                    const v16h* __restrict__ WeF,
                    const v16h* __restrict__ WaF,
                    const float* __restrict__ be, const float* __restrict__ ba,
                    float* __restrict__ ws_w,
                    float* __restrict__ ws_e,
                    float* __restrict__ ws_a)
{
    __shared__ float lds_logits[16][64];

    const int lane = threadIdx.x;
    const int m    = lane & 15;
    const int hi   = lane >> 4;
    const int row0 = blockIdx.x * 16;
    const int bt   = row0 + m;

    const int cs = concept_seq[bt];
    const int cr = correct_seq[bt];
    const v8h* kp = (const v8h*)(ek_h + (size_t)cs * D_);
    const v8h* vp = (const v8h*)(ev_h + (size_t)(cs + NC_ * cr) * D_);

    v16h ka[2], va[2];
#pragma unroll
    for (int c = 0; c < 2; ++c) {
        ka[c] = cat8(kp[c * 4 + hi], kp[c * 4 + 2 + hi]);
        va[c] = cat8(vp[c * 4 + hi], vp[c * 4 + 2 + hi]);
    }

    // ---- logits = k @ Mk^T ----
#pragma unroll
    for (int nt = 0; nt < 4; ++nt) {
        v8f acc = {};
#pragma unroll
        for (int c = 0; c < 2; ++c)
            acc = wmma_f16(ka[c], MkF[(nt * 2 + c) * 32 + lane], acc);
#pragma unroll
        for (int i = 0; i < 8; ++i) lds_logits[i + hi * 8][nt * 16 + m] = acc[i];
    }
    __syncthreads();

    // ---- softmax over 50 slots, lanes 0..15 handle one row each ----
    if (hi == 0) {
        float mx = -1e30f;
        for (int s = 0; s < S_; ++s) mx = fmaxf(mx, lds_logits[m][s]);
        float sum = 0.f;
        for (int s = 0; s < S_; ++s) sum += __expf(lds_logits[m][s] - mx);
        const float inv = 1.f / sum;
        for (int s = 0; s < S_; ++s)
            ws_w[(size_t)bt * 64 + s] = __expf(lds_logits[m][s] - mx) * inv;
    }

    // ---- e = sigmoid(v@We+be), a = tanh(v@Wa+ba) ----
#pragma unroll
    for (int nt = 0; nt < 4; ++nt) {
        v8f eacc = {}, aacc = {};
#pragma unroll
        for (int c = 0; c < 2; ++c) {
            eacc = wmma_f16(va[c], WeF[(nt * 2 + c) * 32 + lane], eacc);
            aacc = wmma_f16(va[c], WaF[(nt * 2 + c) * 32 + lane], aacc);
        }
        const int n = nt * 16 + m;
        const float bbe = be[n], bba = ba[n];
#pragma unroll
        for (int i = 0; i < 8; ++i) {
            const size_t row = (size_t)(row0 + i + hi * 8);
            ws_e[row * 64 + n] = 1.f / (1.f + __expf(-(eacc[i] + bbe)));
            ws_a[row * 64 + n] = tanhf(aacc[i] + bba);
        }
    }
}

// -------------------------------------------------------------------------
// Phase 2: sequential scan, one block per batch element, Mv in LDS (f32).
// Thread (g,d) owns Mv[s0..s1)[d]; only the r partial sum needs barriers.
// -------------------------------------------------------------------------
__global__ __launch_bounds__(256)
void deepirt_phase2(const float* __restrict__ Mv0,
                    const float* __restrict__ ws_w,
                    const float* __restrict__ ws_e,
                    const float* __restrict__ ws_a,
                    float* __restrict__ ws_r)
{
    __shared__ float Mv[S_ * D_];
    __shared__ float rp[4][D_];

    const int b   = blockIdx.x;
    const int tid = threadIdx.x;
    const int d   = tid & 63;
    const int g   = tid >> 6;

    for (int i = tid; i < S_ * D_; i += 256) Mv[i] = Mv0[i];
    __syncthreads();

    const int s0 = (g * S_) / 4;
    const int s1 = ((g + 1) * S_) / 4;

    for (int t = 0; t < T_; ++t) {
        const size_t bt = (size_t)b * T_ + t;
        const float ev = ws_e[bt * 64 + d];
        const float av = ws_a[bt * 64 + d];
        float racc = 0.f;
        for (int s = s0; s < s1; ++s) {
            const float wv = ws_w[bt * 64 + s];
            const float mv = Mv[s * D_ + d];     // Mv_pre
            racc += wv * mv;
            Mv[s * D_ + d] = mv * (1.f - wv * ev) + wv * av;
        }
        rp[g][d] = racc;
        __syncthreads();
        if (g == 0)
            ws_r[bt * 64 + d] = rp[0][d] + rp[1][d] + rp[2][d] + rp[3][d];
        __syncthreads();
    }
}

// -------------------------------------------------------------------------
// Phase 3: f = tanh([r | k] @ Wf + bf) (K=128), then both heads + sigmoid.
// -------------------------------------------------------------------------
__global__ __launch_bounds__(32)
void deepirt_phase3(const int*   __restrict__ concept_seq,
                    const float* __restrict__ embed_key,
                    const _Float16* __restrict__ ek_h,
                    const float* __restrict__ ws_r,
                    const v16h* __restrict__ WfF, const float* __restrict__ bf,
                    const float* __restrict__ Wab, const float* __restrict__ bab,
                    const float* __restrict__ Wd,  const float* __restrict__ bd,
                    float* __restrict__ out)
{
    __shared__ float fsh[16][64];
    __shared__ float ksh[16][64];

    const int lane = threadIdx.x;
    const int m    = lane & 15;
    const int hi   = lane >> 4;
    const int row0 = blockIdx.x * 16;
    const int bt   = row0 + m;

    const int   cs   = concept_seq[bt];
    const float* rrow = ws_r + (size_t)bt * 64;
    const float* krow = embed_key + (size_t)cs * D_;
    const v8h*   kp   = (const v8h*)(ek_h + (size_t)cs * D_);

    // A fragments: K-chunks 0,1 from r (f32 -> f16); 2,3 from f16 key table
    v16h afrag[4];
#pragma unroll
    for (int c = 0; c < 2; ++c) {
        const float* src = rrow + c * 32;
#pragma unroll
        for (int j = 0; j < 16; ++j) {
            const int kk = (j < 8 ? hi * 8 + j : 16 + hi * 8 + (j - 8));
            afrag[c][j] = (_Float16)src[kk];
        }
        afrag[2 + c] = cat8(kp[c * 4 + hi], kp[c * 4 + 2 + hi]);
    }

    // stage k rows (f32) for the difficulty head
#pragma unroll
    for (int j = 0; j < 32; ++j) ksh[m][hi * 32 + j] = krow[hi * 32 + j];

#pragma unroll
    for (int nt = 0; nt < 4; ++nt) {
        v8f acc = {};
#pragma unroll
        for (int c = 0; c < 4; ++c)
            acc = wmma_f16(afrag[c], WfF[(nt * 4 + c) * 32 + lane], acc);
        const int n = nt * 16 + m;
        const float bb = bf[n];
#pragma unroll
        for (int i = 0; i < 8; ++i) fsh[i + hi * 8][n] = tanhf(acc[i] + bb);
    }
    __syncthreads();

    if (hi == 0) {
        float sab = bab[0];
        float sd  = bd[0];
        for (int j = 0; j < D_; ++j) {
            sab += fsh[m][j] * Wab[j];
            sd  += ksh[m][j] * Wd[j];
        }
        const float ability = tanhf(sab);
        const float diff    = tanhf(sd);
        out[bt] = 1.f / (1.f + __expf(-(3.f * ability - diff)));
    }
}

// -------------------------------------------------------------------------
extern "C" void kernel_launch(void* const* d_in, const int* in_sizes, int n_in,
                              void* d_out, int out_size, void* d_ws, size_t ws_size,
                              hipStream_t stream)
{
    const int*   concept_seq = (const int*)  d_in[0];
    const int*   correct_seq = (const int*)  d_in[1];
    // d_in[2] = num_concept scalar (== 1024, baked in as NC_)
    const float* embed_key   = (const float*)d_in[3];
    const float* embed_value = (const float*)d_in[4];
    const float* Mk   = (const float*)d_in[5];
    const float* Mv0  = (const float*)d_in[6];
    const float* Wf   = (const float*)d_in[7];
    const float* bf   = (const float*)d_in[8];
    const float* We   = (const float*)d_in[9];
    const float* be   = (const float*)d_in[10];
    const float* Wa   = (const float*)d_in[11];
    const float* ba   = (const float*)d_in[12];
    const float* Wab  = (const float*)d_in[13];
    const float* bab  = (const float*)d_in[14];
    const float* Wd   = (const float*)d_in[15];
    const float* bd   = (const float*)d_in[16];

    // f32 workspace: 4 arrays of BT*64 (w, e, a, r) = 32.8 MB
    float* ws   = (float*)d_ws;
    float* ws_w = ws;
    float* ws_e = ws + (size_t)BT_ * 64;
    float* ws_a = ws + (size_t)BT_ * 64 * 2;
    float* ws_r = ws + (size_t)BT_ * 64 * 3;

    // f16 workspace region (~0.43 MB): converted tables + packed B fragments
    _Float16* hb   = (_Float16*)(ws + (size_t)4 * BT_ * 64);
    _Float16* ek_h = hb;
    _Float16* ev_h = ek_h + EKH_SZ;
    _Float16* MkF  = ev_h + EVH_SZ;
    _Float16* WeF  = MkF + FRAG8_SZ;
    _Float16* WaF  = WeF + FRAG8_SZ;
    _Float16* WfF  = WaF + FRAG8_SZ;

    const int cvt_elems  = EKH_SZ + EVH_SZ;                 // 196608
    const int pack_elems = 3 * FRAG8_SZ + FRAG16_SZ;        // 20480

    deepirt_cvt_tables<<<(cvt_elems + 255) / 256, 256, 0, stream>>>(
        embed_key, embed_value, ek_h, ev_h);

    deepirt_pack_frags<<<(pack_elems + 255) / 256, 256, 0, stream>>>(
        Mk, We, Wa, Wf, MkF, WeF, WaF, WfF);

    deepirt_phase1<<<BT_ / 16, 32, 0, stream>>>(
        concept_seq, correct_seq, ek_h, ev_h,
        (const v16h*)MkF, (const v16h*)WeF, (const v16h*)WaF,
        be, ba, ws_w, ws_e, ws_a);

    deepirt_phase2<<<B_, 256, 0, stream>>>(Mv0, ws_w, ws_e, ws_a, ws_r);

    deepirt_phase3<<<BT_ / 16, 32, 0, stream>>>(
        concept_seq, embed_key, ek_h, ws_r,
        (const v16h*)WfF, bf, Wab, bab, Wd, bd, (float*)d_out);
}
